// CausalMultiHeadSelfAttention_39573828665745
// MI455X (gfx1250) — compile-verified, matrix operand order checked by reading, not run on hardware
//
#include <hip/hip_runtime.h>
#include <hip/hip_bf16.h>

// ---------------------------------------------------------------------------
// Causal MHA with RoPE for MI455X (gfx1250, wave32, WMMA bf16 16x16x32).
//   B=4, S=2048, D_MODEL=1024, H=16, D_K=64
// Pipeline: f32->bf16 convert | QKV WMMA GEMM (+fused RoPE, Q scaled) |
//           flash attention, transposed-score form, 32 queries/wave,
//           full/diagonal key-block split, spill-free register budget |
//           output WMMA GEMM -> f32.
// ---------------------------------------------------------------------------

#define BATCH   4
#define SEQ     2048
#define DMODEL  1024
#define NHEADS  16
#define DK      64

typedef __attribute__((ext_vector_type(16))) __bf16 v16bf;
typedef __attribute__((ext_vector_type(8)))  __bf16 v8bf;
typedef __attribute__((ext_vector_type(8)))  float  v8f;

union BF16x16 { v16bf v; v8bf h[2]; };

// f32 -> bf16: native conversion (backend emits v_cvt_*bf16* when available,
// falls back to inline RNE expansion otherwise — compile-safe either way).
__device__ __forceinline__ __bf16 f2bf(float f) {
    return (__bf16)f;
}

__device__ __forceinline__ unsigned pack2bf(float lo, float hi) {
    union { __bf16 b[2]; unsigned u; } r;
    r.b[0] = (__bf16)lo; r.b[1] = (__bf16)hi;
    return r.u;
}

__device__ __forceinline__ v8f bfmma(v16bf a, v16bf b, v8f c) {
    // D = A(16x32 bf16) x B(32x16 bf16) + C(16x16 f32)
    return __builtin_amdgcn_wmma_f32_16x16x32_bf16(
        false, a, false, b, (short)0, c, false, false);
}

// Load a 16x32 bf16 fragment (A layout; the B layout mirrors it, with
// identical per-lane addressing when B(k,n) is stored row-major by n).
__device__ __forceinline__ v16bf ldfrag(const __bf16* p, int ld, int row0,
                                        int k0, int lane) {
    int r  = row0 + (lane & 15);
    int cb = k0 + ((lane >> 4) << 3);
    const __bf16* base = p + (size_t)r * ld + cb;
    BF16x16 f;
    f.h[0] = *(const v8bf*)(base);
    f.h[1] = *(const v8bf*)(base + 16);
    return f.v;
}

// ---------------------------------------------------------------------------
// Kernel 1: f32 -> bf16 conversion
// ---------------------------------------------------------------------------
__global__ void cvt_bf16_kernel(const float* __restrict__ src,
                                __bf16* __restrict__ dst, int n) {
    int i = blockIdx.x * blockDim.x + threadIdx.x;
    if (i < n) dst[i] = f2bf(src[i]);
}

// ---------------------------------------------------------------------------
// Kernel 2: WMMA GEMM  Y[m,n] = sum_k A[m,k] * W[n,k]
// One wave computes a 32x32 tile (2x2 frags). 128 thr/block = 4 waves.
// MODE 0: Q  -> RoPE, scale 1/8, bf16 [B,H,S,DK]
// MODE 1: K  -> RoPE,            bf16 [B,H,S,DK]
// MODE 2: V  ->                  bf16 [B,H,DK,S]  (transposed for PV A-frags)
// MODE 3: out-proj ->            f32  [B,S,D] (d_out)
// ---------------------------------------------------------------------------
template <int MODE>
__device__ __forceinline__ void gemm_epilogue(v8f c, int mr, int nc,
                                              void* out,
                                              const int* __restrict__ tokpos,
                                              int lane) {
    const int half = lane >> 4;
    const int col  = nc + (lane & 15);
    if (MODE <= 1) {
        const int d    = col & (DK - 1);
        const float freq = __powf(10000.0f, -(float)(d & ~1) / (float)DK);
        const float sgn  = (d & 1) ? 1.0f : -1.0f;
        const float scl  = (MODE == 0) ? 0.125f : 1.0f;  // 1/sqrt(DK) folded into Q
        const int h = col >> 6;
        __bf16* O = (__bf16*)out;
#pragma unroll
        for (int j = 0; j < 8; ++j) {
            int row = mr + j + 8 * half;      // flat b*S + s
            int s   = row & (SEQ - 1);
            int b   = row >> 11;
            float ang = (float)tokpos[s] * freq;
            float sn, cs;
            __sincosf(ang, &sn, &cs);
            float part = __shfl_xor(c[j], 1, 32);    // pair partner (col^1)
            float val  = (c[j] * cs + sgn * part * sn) * scl;
            O[(((size_t)(b * NHEADS + h) * SEQ + s) * DK) + d] = f2bf(val);
        }
    } else if (MODE == 2) {
        const int h = col >> 6;
        const int d = col & (DK - 1);
        __bf16* O = (__bf16*)out;
#pragma unroll
        for (int j = 0; j < 8; ++j) {
            int row = mr + j + 8 * half;
            int s   = row & (SEQ - 1);
            int b   = row >> 11;
            O[(((size_t)(b * NHEADS + h) * DK + d) * SEQ) + s] = f2bf(c[j]);
        }
    } else {
        float* O = (float*)out;
#pragma unroll
        for (int j = 0; j < 8; ++j) {
            int row = mr + j + 8 * half;
            O[(size_t)row * DMODEL + col] = c[j];
        }
    }
}

template <int MODE>
__global__ void __launch_bounds__(128, 1)
proj_gemm_kernel(const __bf16* __restrict__ A,
                 const __bf16* __restrict__ W,
                 void* __restrict__ out,
                 const int* __restrict__ tokpos) {
    const int lane = threadIdx.x & 31;
    const int wave = threadIdx.x >> 5;
    const int tile = blockIdx.x * 4 + wave;     // 256*32 = 8192 tiles
    const int m0 = (tile >> 5) * 32;            // 0..8160
    const int n0 = (tile & 31) * 32;            // 0..992

    v8f c00 = {}, c01 = {}, c10 = {}, c11 = {};
#pragma unroll 4
    for (int k = 0; k < DMODEL; k += 32) {
        v16bf a0 = ldfrag(A, DMODEL, m0,      k, lane);
        v16bf a1 = ldfrag(A, DMODEL, m0 + 16, k, lane);
        v16bf b0 = ldfrag(W, DMODEL, n0,      k, lane);
        v16bf b1 = ldfrag(W, DMODEL, n0 + 16, k, lane);
        if (k + 32 < DMODEL) {   // uniform; exercises global_prefetch path
            __builtin_prefetch(W + (size_t)(n0 + (lane & 15)) * DMODEL + k + 32, 0, 3);
        }
        c00 = bfmma(a0, b0, c00);
        c01 = bfmma(a0, b1, c01);
        c10 = bfmma(a1, b0, c10);
        c11 = bfmma(a1, b1, c11);
    }
    gemm_epilogue<MODE>(c00, m0,      n0,      out, tokpos, lane);
    gemm_epilogue<MODE>(c01, m0,      n0 + 16, out, tokpos, lane);
    gemm_epilogue<MODE>(c10, m0 + 16, n0,      out, tokpos, lane);
    gemm_epilogue<MODE>(c11, m0 + 16, n0 + 16, out, tokpos, lane);
}

// ---------------------------------------------------------------------------
// Kernel 3: flash attention (causal), transposed-score formulation.
// One wave = 32 queries (two 16-wide column tiles) of one (b,h).
// Column tiles processed sequentially through one pair of score frags to cap
// the live register set (spill-free with __launch_bounds__(128,1)).
// ---------------------------------------------------------------------------

// Online-softmax update for one column tile. sa = keys[kbase..], sb = +16.
// Rescales o[0..3], fills pb with bf16 probabilities in B-frag order.
// P-values are converted pairwise (adjacent pb elements share a VGPR) so the
// backend can fuse into packed bf16 converts.
__device__ __forceinline__ void softmax_update(v8f& sa, v8f& sb,
                                               float& m, float& l,
                                               v8f* o, BF16x16& pb) {
    float mx = -3.0e38f;
#pragma unroll
    for (int j = 0; j < 8; ++j) mx = fmaxf(mx, fmaxf(sa[j], sb[j]));
    mx = fmaxf(mx, __shfl_xor(mx, 16, 32));    // merge the two key halves
    const float mn    = fmaxf(m, mx);
    const float alpha = __expf(m - mn);
    m = mn;
    float p0[8], p1[8];
    float rs = 0.0f;
#pragma unroll
    for (int j = 0; j < 8; ++j) {
        p0[j] = __expf(sa[j] - mn);
        p1[j] = __expf(sb[j] - mn);
        rs += p0[j] + p1[j];
    }
#pragma unroll
    for (int j = 0; j < 8; j += 2) {           // pairwise -> packed converts
        pb.v[j]     = f2bf(p0[j]);   pb.v[j + 1] = f2bf(p0[j + 1]);
        pb.v[j + 8] = f2bf(p1[j]);   pb.v[j + 9] = f2bf(p1[j + 1]);
    }
    rs += __shfl_xor(rs, 16, 32);
    l = l * alpha + rs;
#pragma unroll
    for (int j = 0; j < 8; ++j) {
        o[0][j] *= alpha; o[1][j] *= alpha; o[2][j] *= alpha; o[3][j] *= alpha;
    }
}

__global__ void __launch_bounds__(128, 1)
attn_kernel(const __bf16* __restrict__ Q,
            const __bf16* __restrict__ K,
            const __bf16* __restrict__ Vt,
            __bf16* __restrict__ O) {
    const int lane = threadIdx.x & 31;
    const int wave = threadIdx.x >> 5;
    const int wid  = blockIdx.x * 4 + wave;         // B*H*(S/32) = 4096
    const int qt   = wid & 63;
    const int bh   = wid >> 6;
    const int q0   = qt * 32;
    const int half = lane >> 4;
    const int qg0  = q0 + (lane & 15);              // query of column tile 0
    const int qg1  = qg0 + 16;                      // query of column tile 1

    const __bf16* Qb = Q  + (size_t)bh * SEQ * DK;
    const __bf16* Kb = K  + (size_t)bh * SEQ * DK;
    const __bf16* Vb = Vt + (size_t)bh * DK * SEQ;

    // B-frags of Q^T for both column tiles (mirror-of-A addressing)
    const v16bf q0_lo = ldfrag(Qb, DK, q0,      0,  lane);
    const v16bf q0_hi = ldfrag(Qb, DK, q0,      32, lane);
    const v16bf q1_lo = ldfrag(Qb, DK, q0 + 16, 0,  lane);
    const v16bf q1_hi = ldfrag(Qb, DK, q0 + 16, 32, lane);

    v8f oc0[4] = {}, oc1[4] = {};                  // O^T accums per column tile
    float m0 = -3.0e38f, l0 = 0.0f;
    float m1 = -3.0e38f, l1 = 0.0f;

    const int nfull = q0 >> 5;                     // fully-unmasked key blocks
    for (int kb = 0; kb <= nfull; ++kb) {          // last iter = diagonal block
        const int j0 = kb * 32;
        const bool diag = (kb == nfull);
        const v16bf ka_lo = ldfrag(Kb, DK, j0,      0,  lane);
        const v16bf ka_hi = ldfrag(Kb, DK, j0,      32, lane);
        const v16bf kc_lo = ldfrag(Kb, DK, j0 + 16, 0,  lane);
        const v16bf kc_hi = ldfrag(Kb, DK, j0 + 16, 32, lane);
        const int kbase = j0 + 8 * half;

        // ---- column tile 0: scores -> softmax -> pb0 (frees score frags) ----
        BF16x16 pb0, pb1;
        {
            v8f sa = {}, sb = {};
            sa = bfmma(ka_lo, q0_lo, sa);  sa = bfmma(ka_hi, q0_hi, sa);
            sb = bfmma(kc_lo, q0_lo, sb);  sb = bfmma(kc_hi, q0_hi, sb);
            if (diag) {
#pragma unroll
                for (int j = 0; j < 8; ++j) {
                    sa[j] = (kbase + j      <= qg0) ? sa[j] : -3.0e38f;
                    sb[j] = (kbase + j + 16 <= qg0) ? sb[j] : -3.0e38f;
                }
            }
            softmax_update(sa, sb, m0, l0, oc0, pb0);
        }
        // ---- column tile 1 ----
        {
            v8f sa = {}, sb = {};
            sa = bfmma(ka_lo, q1_lo, sa);  sa = bfmma(ka_hi, q1_hi, sa);
            sb = bfmma(kc_lo, q1_lo, sb);  sb = bfmma(kc_hi, q1_hi, sb);
            if (diag) {
#pragma unroll
                for (int j = 0; j < 8; ++j) {
                    sa[j] = (kbase + j      <= qg1) ? sa[j] : -3.0e38f;
                    sb[j] = (kbase + j + 16 <= qg1) ? sb[j] : -3.0e38f;
                }
            }
            softmax_update(sa, sb, m1, l1, oc1, pb1);
        }

        // O^T += V^T * P^T  (A = V^T rows of d, contiguous along keys)
#pragma unroll
        for (int t = 0; t < 4; ++t) {
            const v16bf vf = ldfrag(Vb, SEQ, t * 16, j0, lane);
            oc0[t] = bfmma(vf, pb0.v, oc0[t]);
            oc1[t] = bfmma(vf, pb1.v, oc1[t]);
        }
    }

    // finalize both column tiles: /= l, write bf16 [B,S,H*DK] packed pairs
    const int b = bh >> 4;
    const int h = bh & 15;
    const float inv0 = 1.0f / l0;
    const float inv1 = 1.0f / l1;
    __bf16* d0 = O + ((size_t)(b * SEQ + qg0)) * DMODEL + h * DK + 8 * half;
    __bf16* d1 = O + ((size_t)(b * SEQ + qg1)) * DMODEL + h * DK + 8 * half;
#pragma unroll
    for (int t = 0; t < 4; ++t) {
#pragma unroll
        for (int j = 0; j < 8; j += 2) {   // d = t*16 + j + 8*half
            *(unsigned*)(d0 + t * 16 + j) = pack2bf(oc0[t][j] * inv0,
                                                    oc0[t][j + 1] * inv0);
            *(unsigned*)(d1 + t * 16 + j) = pack2bf(oc1[t][j] * inv1,
                                                    oc1[t][j + 1] * inv1);
        }
    }
}

// ---------------------------------------------------------------------------
// Host-side launch
// ---------------------------------------------------------------------------
extern "C" void kernel_launch(void* const* d_in, const int* in_sizes, int n_in,
                              void* d_out, int out_size, void* d_ws, size_t ws_size,
                              hipStream_t stream) {
    const float* x   = (const float*)d_in[0];
    const float* wq  = (const float*)d_in[1];
    const float* wk  = (const float*)d_in[2];
    const float* wv  = (const float*)d_in[3];
    const float* wo  = (const float*)d_in[4];
    const int*   pos = (const int*)d_in[5];

    const size_t MB = 1024 * 1024;
    char* ws = (char*)d_ws;
    __bf16* Xbf  = (__bf16*)(ws);                 // 16 MB  (8192x1024)
    __bf16* Wqb  = (__bf16*)(ws + 16 * MB);       //  2 MB
    __bf16* Wkb  = (__bf16*)(ws + 18 * MB);       //  2 MB
    __bf16* Wvb  = (__bf16*)(ws + 20 * MB);       //  2 MB
    __bf16* Wob  = (__bf16*)(ws + 22 * MB);       //  2 MB
    __bf16* Qbf  = (__bf16*)(ws + 24 * MB);       // 16 MB [B,H,S,DK]
    __bf16* Kbf  = (__bf16*)(ws + 40 * MB);       // 16 MB [B,H,S,DK]
    __bf16* Vtb  = (__bf16*)(ws + 56 * MB);       // 16 MB [B,H,DK,S]
    __bf16* Obf  = (__bf16*)(ws + 72 * MB);       // 16 MB [B,S,D]

    const int NX = BATCH * SEQ * DMODEL;          // 8388608
    const int NW = DMODEL * DMODEL;               // 1048576

    cvt_bf16_kernel<<<(NX + 255) / 256, 256, 0, stream>>>(x,  Xbf, NX);
    cvt_bf16_kernel<<<(NW + 255) / 256, 256, 0, stream>>>(wq, Wqb, NW);
    cvt_bf16_kernel<<<(NW + 255) / 256, 256, 0, stream>>>(wk, Wkb, NW);
    cvt_bf16_kernel<<<(NW + 255) / 256, 256, 0, stream>>>(wv, Wvb, NW);
    cvt_bf16_kernel<<<(NW + 255) / 256, 256, 0, stream>>>(wo, Wob, NW);

    // 8192 wave tiles / 4 waves per block
    const int gemmBlocks = 2048;
    proj_gemm_kernel<0><<<gemmBlocks, 128, 0, stream>>>(Xbf, Wqb, Qbf, pos);
    proj_gemm_kernel<1><<<gemmBlocks, 128, 0, stream>>>(Xbf, Wkb, Kbf, pos);
    proj_gemm_kernel<2><<<gemmBlocks, 128, 0, stream>>>(Xbf, Wvb, Vtb, pos);

    // B*H*(S/32) = 4096 waves / 4 per block
    attn_kernel<<<1024, 128, 0, stream>>>(Qbf, Kbf, Vtb, Obf);

    proj_gemm_kernel<3><<<gemmBlocks, 128, 0, stream>>>(Obf, Wob, d_out, pos);
}
